// GPRGNN_27152783245355
// MI455X (gfx1250) — matrix operand order repaired and checked by reading, not compiled, not run on hardware
//
#include <hip/hip_runtime.h>
#include <hip/hip_bf16.h>

#define NFEAT 512
#define NHID  256
#define NCLS  64
#define KHOP  10

typedef __bf16 bf16_t;
typedef __attribute__((ext_vector_type(16))) __bf16 v16bf;
typedef __attribute__((ext_vector_type(8)))  __bf16 v8bf;
typedef __attribute__((ext_vector_type(8)))  float  v8f;
typedef __attribute__((ext_vector_type(4)))  float  v4f;

// ---------------------------------------------------------------------------
// degree / normalization
// ---------------------------------------------------------------------------
__global__ void gpr_init_deg(float* deg, int N) {
    int i = blockIdx.x * blockDim.x + threadIdx.x;
    if (i < N) deg[i] = 1.0f;                       // self-loop
}

__global__ void gpr_count_deg(const int* __restrict__ col, float* deg, int E) {
    int e = blockIdx.x * blockDim.x + threadIdx.x;
    if (e < E) atomicAdd(&deg[col[e]], 1.0f);
}

__global__ void gpr_make_dis(float* deg, int N) {
    int i = blockIdx.x * blockDim.x + threadIdx.x;
    if (i < N) {
        float d = deg[i];
        deg[i] = (d > 0.0f) ? rsqrtf(d) : 0.0f;     // in place: deg -> dis
    }
}

// ---------------------------------------------------------------------------
// GEMM1: H1[N,256] = bf16( relu( X[N,512] @ W1[512,256] + b1 ) )
// block tile 128x128, 8 waves (4 row groups x 2 col groups), BK=32.
// Register double-buffering: tile k+1's global loads are issued right after
// the barrier and consumed at the top of the next iteration, so they fly
// under the WMMA + ds_load stage (no serialized load->wait->store chains).
// OOB rows are clamped to N-1 (results discarded in epilogue) => branch-free.
// ---------------------------------------------------------------------------
__global__ __launch_bounds__(256) void gpr_gemm1_wmma(
    const float* __restrict__ X, const float* __restrict__ W1,
    const float* __restrict__ b1, bf16_t* __restrict__ H1, int N)
{
    __shared__ bf16_t As[128 * 40];                 // row-major [row][k], stride 40
    __shared__ bf16_t Bs[128 * 40];                 // N-major  [n][k],   stride 40

    const int tid  = threadIdx.x;
    const int lane = tid & 31;
    const int wave = tid >> 5;
    const int wm   = wave & 3;                      // row group (32 rows each)
    const int wn   = wave >> 2;                     // col group (64 cols each)
    const int r0   = blockIdx.x * 128;
    const int c0   = blockIdx.y * 128;
    const int half = lane >> 4;
    const int mr   = lane & 15;

    v8f acc[2][4];
    const v8f zero8 = {0.f,0.f,0.f,0.f,0.f,0.f,0.f,0.f};
#pragma unroll
    for (int mf = 0; mf < 2; ++mf)
#pragma unroll
        for (int nf = 0; nf < 4; ++nf) acc[mf][nf] = zero8;

    v4f aReg[4], bReg[4];

    auto issueLoads = [&](int kk) {
#pragma unroll
        for (int i = 0; i < 4; ++i) {               // A: 128x32, 8 float4/row
            int idx = tid + i * 256;
            int row = idx >> 3, q = idx & 7;
            int grow = r0 + row;
            if (grow >= N) grow = N - 1;            // clamped, discarded later
            aReg[i] = *(const v4f*)(X + (size_t)grow * NFEAT + kk + q * 4);
        }
#pragma unroll
        for (int i = 0; i < 4; ++i) {               // B: 32x128, 32 float4/k-row
            int idx = tid + i * 256;
            int k = idx >> 5, q = idx & 31;
            bReg[i] = *(const v4f*)(W1 + (size_t)(kk + k) * NHID + c0 + q * 4);
        }
    };

    auto stageLDS = [&]() {
#pragma unroll
        for (int i = 0; i < 4; ++i) {
            int idx = tid + i * 256;
            int row = idx >> 3, q = idx & 7;
            bf16_t* dst = &As[row * 40 + q * 4];
            dst[0] = (bf16_t)aReg[i][0]; dst[1] = (bf16_t)aReg[i][1];
            dst[2] = (bf16_t)aReg[i][2]; dst[3] = (bf16_t)aReg[i][3];
        }
#pragma unroll
        for (int i = 0; i < 4; ++i) {
            int idx = tid + i * 256;
            int k = idx >> 5, q = idx & 31;
#pragma unroll
            for (int j = 0; j < 4; ++j) Bs[(q * 4 + j) * 40 + k] = (bf16_t)bReg[i][j];
        }
    };

    issueLoads(0);
    for (int kk = 0; kk < NFEAT; kk += 32) {
        stageLDS();                                 // waits the in-flight batch
        __syncthreads();
        if (kk + 32 < NFEAT) issueLoads(kk + 32);   // next tile, not waited

        // --- fragments per documented CDNA5 16-bit layouts ---
        v16bf afr[2], bfr[4];
#pragma unroll
        for (int mf = 0; mf < 2; ++mf) {
            const bf16_t* ap = &As[(wm * 32 + mf * 16 + mr) * 40 + half * 8];
            v8bf a0 = *(const v8bf*)ap;             // K = half*8 .. +7
            v8bf a1 = *(const v8bf*)(ap + 16);      // K = 16+half*8 .. +7
#pragma unroll
            for (int i = 0; i < 8; ++i) { afr[mf][i] = a0[i]; afr[mf][i + 8] = a1[i]; }
        }
#pragma unroll
        for (int nf = 0; nf < 4; ++nf) {
            const bf16_t* bp = &Bs[(wn * 64 + nf * 16 + mr) * 40 + half * 8];
            v8bf b0 = *(const v8bf*)bp;
            v8bf b1 = *(const v8bf*)(bp + 16);
#pragma unroll
            for (int i = 0; i < 8; ++i) { bfr[nf][i] = b0[i]; bfr[nf][i + 8] = b1[i]; }
        }
#pragma unroll
        for (int mf = 0; mf < 2; ++mf)
#pragma unroll
            for (int nf = 0; nf < 4; ++nf)
                acc[mf][nf] = __builtin_amdgcn_wmma_f32_16x16x32_bf16(
                    false, afr[mf], false, bfr[nf], (short)0, acc[mf][nf],
                    false, false);
        __syncthreads();
    }

    // --- epilogue: bias + relu + bf16 store ---
#pragma unroll
    for (int mf = 0; mf < 2; ++mf)
#pragma unroll
        for (int nf = 0; nf < 4; ++nf) {
            int gcol = c0 + wn * 64 + nf * 16 + mr;
            float bias = b1[gcol];
#pragma unroll
            for (int r = 0; r < 8; ++r) {
                int grow = r0 + wm * 32 + mf * 16 + half * 8 + r;
                if (grow < N) {
                    float v = acc[mf][nf][r] + bias;
                    v = fmaxf(v, 0.0f);
                    H1[(size_t)grow * NHID + gcol] = (bf16_t)v;
                }
            }
        }
}

// ---------------------------------------------------------------------------
// GEMM2: Hc[N,64] = H1[N,256] @ W2[256,64] + b2 ; Hid = temp[0]*Hc
// block tile 128x64, 8 waves (4 row groups x 2 col groups of 32), BK=32
// Same register double-buffering scheme as GEMM1.
// ---------------------------------------------------------------------------
__global__ __launch_bounds__(256) void gpr_gemm2_wmma(
    const bf16_t* __restrict__ H1, const float* __restrict__ W2,
    const float* __restrict__ b2, const float* __restrict__ temp,
    float* __restrict__ Hc, float* __restrict__ Hid, int N)
{
    __shared__ bf16_t As[128 * 40];
    __shared__ bf16_t Bs[64 * 40];

    const int tid  = threadIdx.x;
    const int lane = tid & 31;
    const int wave = tid >> 5;
    const int wm   = wave & 3;
    const int wn   = wave >> 2;                     // 2 groups of 32 cols
    const int r0   = blockIdx.x * 128;
    const int half = lane >> 4;
    const int mr   = lane & 15;

    v8f acc[2][2];
    const v8f zero8 = {0.f,0.f,0.f,0.f,0.f,0.f,0.f,0.f};
#pragma unroll
    for (int mf = 0; mf < 2; ++mf)
#pragma unroll
        for (int nf = 0; nf < 2; ++nf) acc[mf][nf] = zero8;

    v8bf aReg[2];
    v4f  bReg[2];

    auto issueLoads = [&](int kk) {
#pragma unroll
        for (int i = 0; i < 2; ++i) {               // A: 128x32 bf16, 4 v8bf/row
            int idx = tid + i * 256;
            int row = idx >> 2, q = idx & 3;
            int grow = r0 + row;
            if (grow >= N) grow = N - 1;
            aReg[i] = *(const v8bf*)(H1 + (size_t)grow * NHID + kk + q * 8);
        }
#pragma unroll
        for (int i = 0; i < 2; ++i) {               // B: 32x64, 16 float4/k-row
            int idx = tid + i * 256;
            int k = idx >> 4, q = idx & 15;
            bReg[i] = *(const v4f*)(W2 + (size_t)(kk + k) * NCLS + q * 4);
        }
    };

    auto stageLDS = [&]() {
#pragma unroll
        for (int i = 0; i < 2; ++i) {
            int idx = tid + i * 256;
            int row = idx >> 2, q = idx & 3;
            *(v8bf*)&As[row * 40 + q * 8] = aReg[i];
        }
#pragma unroll
        for (int i = 0; i < 2; ++i) {
            int idx = tid + i * 256;
            int k = idx >> 4, q = idx & 15;
#pragma unroll
            for (int j = 0; j < 4; ++j) Bs[(q * 4 + j) * 40 + k] = (bf16_t)bReg[i][j];
        }
    };

    issueLoads(0);
    for (int kk = 0; kk < NHID; kk += 32) {
        stageLDS();
        __syncthreads();
        if (kk + 32 < NHID) issueLoads(kk + 32);

        v16bf afr[2], bfr[2];
#pragma unroll
        for (int mf = 0; mf < 2; ++mf) {
            const bf16_t* ap = &As[(wm * 32 + mf * 16 + mr) * 40 + half * 8];
            v8bf a0 = *(const v8bf*)ap;
            v8bf a1 = *(const v8bf*)(ap + 16);
#pragma unroll
            for (int i = 0; i < 8; ++i) { afr[mf][i] = a0[i]; afr[mf][i + 8] = a1[i]; }
        }
#pragma unroll
        for (int nf = 0; nf < 2; ++nf) {
            const bf16_t* bp = &Bs[(wn * 32 + nf * 16 + mr) * 40 + half * 8];
            v8bf b0 = *(const v8bf*)bp;
            v8bf b1 = *(const v8bf*)(bp + 16);
#pragma unroll
            for (int i = 0; i < 8; ++i) { bfr[nf][i] = b0[i]; bfr[nf][i + 8] = b1[i]; }
        }
#pragma unroll
        for (int mf = 0; mf < 2; ++mf)
#pragma unroll
            for (int nf = 0; nf < 2; ++nf)
                acc[mf][nf] = __builtin_amdgcn_wmma_f32_16x16x32_bf16(
                    false, afr[mf], false, bfr[nf], (short)0, acc[mf][nf],
                    false, false);
        __syncthreads();
    }

    const float t0 = temp[0];
#pragma unroll
    for (int mf = 0; mf < 2; ++mf)
#pragma unroll
        for (int nf = 0; nf < 2; ++nf) {
            int gcol = wn * 32 + nf * 16 + mr;
            float bias = b2[gcol];
#pragma unroll
            for (int r = 0; r < 8; ++r) {
                int grow = r0 + wm * 32 + mf * 16 + half * 8 + r;
                if (grow < N) {
                    float v = acc[mf][nf][r] + bias;
                    size_t o = (size_t)grow * NCLS + gcol;
                    Hc[o]  = v;
                    Hid[o] = t0 * v;
                }
            }
        }
}

// ---------------------------------------------------------------------------
// propagation: h_next = D^-1/2 (A+I) D^-1/2 h_cur   (h fits in 192MB L2)
// ---------------------------------------------------------------------------
__global__ void gpr_prop_self(const float* __restrict__ dis,
                              const float* __restrict__ hcur,
                              float* __restrict__ hnext, int N)
{
    int id = blockIdx.x * blockDim.x + threadIdx.x;      // N*64 elements
    if (id < N * NCLS) {
        int i = id >> 6;
        float d = dis[i];
        hnext[id] = d * d * hcur[id];                    // self-loop term
    }
}

// one wave per edge: e = tid>>5 is wave-uniform -> row/col loads scalarize;
// each lane handles features f and f+32 (two coalesced 128B segments)
__global__ void gpr_prop_edge(const int* __restrict__ row,
                              const int* __restrict__ col,
                              const float* __restrict__ dis,
                              const float* __restrict__ hcur,
                              float* __restrict__ hnext, int E)
{
    long long t = (long long)blockIdx.x * blockDim.x + threadIdx.x;
    long long e = t >> 5;
    int f = (int)(t & 31);
    if (e < (long long)E) {
        int r = row[e], c = col[e];
        float w = dis[r] * dis[c];
        const float* src = hcur + (size_t)r * NCLS;
        float*       dst = hnext + (size_t)c * NCLS;
        atomicAdd(&dst[f],      w * src[f]);
        atomicAdd(&dst[f + 32], w * src[f + 32]);
    }
}

__global__ void gpr_axpy(float* __restrict__ hid, const float* __restrict__ h,
                         const float* __restrict__ temp, int k, int N)
{
    int id = blockIdx.x * blockDim.x + threadIdx.x;
    if (id < N * NCLS) hid[id] += temp[k] * h[id];
}

// ---------------------------------------------------------------------------
// log_softmax over 64 classes: one wave per row, 2 values per lane
// ---------------------------------------------------------------------------
__global__ __launch_bounds__(256) void gpr_logsoftmax64(
    const float* __restrict__ hid, float* __restrict__ out, int N)
{
    int wave = threadIdx.x >> 5;
    int lane = threadIdx.x & 31;
    int rowi = blockIdx.x * 8 + wave;
    if (rowi >= N) return;
    const float* hr = hid + (size_t)rowi * NCLS;
    float a = hr[lane], b = hr[lane + 32];
    float m = fmaxf(a, b);
#pragma unroll
    for (int off = 16; off; off >>= 1) m = fmaxf(m, __shfl_xor(m, off, 32));
    float s = __expf(a - m) + __expf(b - m);
#pragma unroll
    for (int off = 16; off; off >>= 1) s += __shfl_xor(s, off, 32);
    float lse = m + __logf(s);
    out[(size_t)rowi * NCLS + lane]      = a - lse;
    out[(size_t)rowi * NCLS + lane + 32] = b - lse;
}

// ---------------------------------------------------------------------------
static inline size_t alignup(size_t x) { return (x + 255) & ~(size_t)255; }

extern "C" void kernel_launch(void* const* d_in, const int* in_sizes, int n_in,
                              void* d_out, int out_size, void* d_ws, size_t ws_size,
                              hipStream_t stream)
{
    const float* x    = (const float*)d_in[0];
    const int*   ei   = (const int*)  d_in[1];
    const float* W1   = (const float*)d_in[2];
    const float* b1   = (const float*)d_in[3];
    const float* W2   = (const float*)d_in[4];
    const float* b2   = (const float*)d_in[5];
    const float* temp = (const float*)d_in[6];

    const int N = in_sizes[0] / NFEAT;
    const int E = in_sizes[1] / 2;
    const int* rowp = ei;
    const int* colp = ei + E;

    char* ws = (char*)d_ws;
    float*  dis = (float*)ws;  ws += alignup((size_t)N * sizeof(float));
    bf16_t* H1  = (bf16_t*)ws; ws += alignup((size_t)N * NHID * sizeof(bf16_t));
    float*  hA  = (float*)ws;  ws += alignup((size_t)N * NCLS * sizeof(float));
    float*  hB  = (float*)ws;  ws += alignup((size_t)N * NCLS * sizeof(float));
    float*  hid = (float*)ws;  ws += alignup((size_t)N * NCLS * sizeof(float));
    (void)ws_size; (void)n_in; (void)out_size;

    const int T = 256;
    // normalization
    gpr_init_deg <<<(N + T - 1) / T, T, 0, stream>>>(dis, N);
    gpr_count_deg<<<(E + T - 1) / T, T, 0, stream>>>(colp, dis, E);
    gpr_make_dis <<<(N + T - 1) / T, T, 0, stream>>>(dis, N);

    // MLP (bf16 WMMA)
    gpr_gemm1_wmma<<<dim3((N + 127) / 128, NHID / 128), T, 0, stream>>>(x, W1, b1, H1, N);
    gpr_gemm2_wmma<<<dim3((N + 127) / 128), T, 0, stream>>>(H1, W2, b2, temp, hA, hid, N);

    // K-hop GPR propagation
    float* cur = hA;
    float* nxt = hB;
    const int nbElem = (N * NCLS + T - 1) / T;
    const int nbEdge = (int)(((long long)E * 32 + T - 1) / T);
    for (int k = 0; k < KHOP; ++k) {
        gpr_prop_self<<<nbElem, T, 0, stream>>>(dis, cur, nxt, N);
        gpr_prop_edge<<<nbEdge, T, 0, stream>>>(rowp, colp, dis, cur, nxt, E);
        gpr_axpy     <<<nbElem, T, 0, stream>>>(hid, nxt, temp, k + 1, N);
        float* t2 = cur; cur = nxt; nxt = t2;
    }

    gpr_logsoftmax64<<<(N + 7) / 8, T, 0, stream>>>(hid, (float*)d_out, N);
}
